// SpatialNetLayer_88691074662594
// MI455X (gfx1250) — matrix operand order repaired and checked by reading, compiled, not run on hardware
//
#include <hip/hip_runtime.h>

// ---------------- problem constants ----------------
#define FQ 129
#define TQ 1000
#define HQ 96
#define GQ 8
#define KQ 5
#define DSq 8
#define DIq 192      // mamba d_inner
#define NST 16       // d_state
#define DRq 6        // dt_rank
#define DCq 4        // mamba conv kernel
#define BQ 2
#define MROWS (BQ*FQ*TQ)          // 258000 rows of H=96 (divisible by 16)
#define XZLD (2*DIq)              // 384
#define XDLD 48                   // padded 38 -> 48 (3 tiles)
#define EPSF 1e-5f

typedef __attribute__((ext_vector_type(16))) __bf16 v16bf;
typedef __attribute__((ext_vector_type(8)))  float  v8f;

__device__ __forceinline__ float siluf(float x) { return x / (1.f + __expf(-x)); }

// ================= Stage 1: fc1 + full_mix + fc2 fused, per (b,t) slice =================
__device__ __forceinline__ void ln_block(const float* xs, float* xn,
                                         const float* g, const float* bb, int tid) {
  if (tid < FQ) {
    const int f = tid;
    float s = 0.f, ss = 0.f;
    for (int h = 0; h < HQ; ++h) { float v = xs[f*HQ+h]; s += v; ss += v*v; }
    float m = s * (1.f/HQ);
    float var = ss * (1.f/HQ) - m*m;
    float r = rsqrtf(var + EPSF);
    for (int h = 0; h < HQ; ++h) xn[f*HQ+h] = (xs[f*HQ+h]-m)*r*g[h] + bb[h];
  }
  __syncthreads();
}

__device__ __forceinline__ void fconv_block(float* xs, float* xn,
    const float* lg, const float* lb, const float* cw, const float* cb,
    const float* pre, int tid) {
  ln_block(xs, xn, lg, lb, tid);
  for (int idx = tid; idx < FQ*HQ; idx += 256) {
    int f = idx / HQ, o = idx % HQ;
    int grp = o / (HQ/GQ);
    const float* wrow = cw + o*(HQ/GQ)*KQ;
    const float* xcol = xn + grp*(HQ/GQ);
    float acc = cb[o];
    for (int ic = 0; ic < HQ/GQ; ++ic)
      for (int k = 0; k < KQ; ++k) {
        int ff = f + k - (KQ/2);
        if (ff >= 0 && ff < FQ) acc += wrow[ic*KQ+k] * xcol[ff*HQ + ic];
      }
    acc = acc >= 0.f ? acc : pre[o]*acc;
    xs[idx] += acc;
  }
  __syncthreads();
}

__global__ __launch_bounds__(256)
void k_stage1(const float* __restrict__ x, float* __restrict__ out,
              const float* f1g, const float* f1b, const float* f1w, const float* f1cb, const float* f1p,
              const float* f2g, const float* f2b, const float* f2w, const float* f2cb, const float* f2p,
              const float* flg, const float* flb,
              const float* sqw, const float* sqb,
              const float* fw,  const float* fb,
              const float* uw,  const float* ub) {
  extern __shared__ float sm[];
  float* xs = sm;                    // F*H current slice (residual stream)
  float* xn = sm + FQ*HQ;            // F*H normalized scratch
  float* yb = xn + FQ*HQ;            // DS*F
  float* zb = yb + DSq*FQ;           // DS*F
  const int tid = threadIdx.x;
  const int bt = blockIdx.x;
  const int b = bt / TQ, t = bt % TQ;
  const long gbase = (long)b*FQ*TQ*HQ + (long)t*HQ;

  for (int idx = tid; idx < FQ*HQ; idx += 256) {
    int f = idx / HQ, h = idx % HQ;
    xs[idx] = x[gbase + (long)f*TQ*HQ + h];
  }
  __syncthreads();

  fconv_block(xs, xn, f1g, f1b, f1w, f1cb, f1p, tid);      // x += fconv1(x)

  // full_mix
  ln_block(xs, xn, flg, flb, tid);
  for (int idx = tid; idx < DSq*FQ; idx += 256) {          // squeeze H->8, silu
    int o = idx / FQ, f = idx % FQ;
    const float* wr = sqw + o*HQ;
    float acc = sqb[o];
    for (int h = 0; h < HQ; ++h) acc += wr[h]*xn[f*HQ+h];
    yb[idx] = siluf(acc);
  }
  __syncthreads();
  for (int idx = tid; idx < DSq*FQ; idx += 256) {          // linear along F
    int o = idx / FQ, gq = idx % FQ;
    const float* wr = fw + gq*FQ;
    const float* yr = yb + o*FQ;
    float acc = fb[gq];
    for (int f = 0; f < FQ; ++f) acc += yr[f]*wr[f];
    zb[o*FQ+gq] = acc;
  }
  __syncthreads();
  for (int idx = tid; idx < FQ*HQ; idx += 256) {           // unsqueeze 8->H, silu, residual
    int f = idx / HQ, h = idx % HQ;
    const float* wr = uw + h*DSq;
    float acc = ub[h];
    for (int o = 0; o < DSq; ++o) acc += wr[o]*zb[o*FQ+f];
    xs[idx] += siluf(acc);
  }
  __syncthreads();

  fconv_block(xs, xn, f2g, f2b, f2w, f2cb, f2p, tid);      // x += fconv2(x)

  for (int idx = tid; idx < FQ*HQ; idx += 256) {
    int f = idx / HQ, h = idx % HQ;
    out[gbase + (long)f*TQ*HQ + h] = xs[idx];
  }
}

// ================= Mamba: per-row LN stats =================
__global__ __launch_bounds__(256)
void k_stats(const float* __restrict__ x, float* __restrict__ mu, float* __restrict__ rs) {
  long row = (long)blockIdx.x*8 + (threadIdx.x >> 5);
  if (row >= MROWS) return;
  int lane = threadIdx.x & 31;
  const float* p = x + row*HQ;
  float v0 = p[lane], v1 = p[lane+32], v2 = p[lane+64];
  float s = v0+v1+v2, ss = v0*v0+v1*v1+v2*v2;
  #pragma unroll
  for (int off = 16; off > 0; off >>= 1) {
    s  += __shfl_xor(s,  off, 32);
    ss += __shfl_xor(ss, off, 32);
  }
  if (lane == 0) {
    float m = s*(1.f/HQ);
    mu[row] = m;
    rs[row] = rsqrtf(ss*(1.f/HQ) - m*m + EPSF);
  }
}

// K-index maps: contiguous k = 16*hi + i, same order for A and B (wave32, bf16 16x16x32)
__device__ __forceinline__ int kmapA(int i, int hi) { return hi*16 + i; }
__device__ __forceinline__ int kmapB(int i, int hi) { return hi*16 + i; }

// ===== in_proj GEMM: xz = LN(x) @ in_w^T  (M=258000, K=96, N=384) =====
// One wave owns 1 M-tile x 4 N-tiles: A fragment (LN fused) reused 4x per k-step.
__global__ __launch_bounds__(256)
void k_inproj(const float* __restrict__ x, const float* __restrict__ mu,
              const float* __restrict__ rs, const float* __restrict__ g,
              const float* __restrict__ bb, const float* __restrict__ W,
              float* __restrict__ xz) {
  const int NT = 4, NGRP = XZLD/16/NT;                    // 6 groups of 4 tiles
  long wid = (long)blockIdx.x*8 + (threadIdx.x >> 5);
  const long total = (long)(MROWS/16)*NGRP;
  if (wid >= total) return;
  const int lane = threadIdx.x & 31, r = lane & 15, hi = lane >> 4;
  const int  gn = (int)(wid % NGRP);
  const long tm = wid / NGRP;
  const long m = tm*16 + r;
  const float mym = mu[m], myr = rs[m];
  const float* xrow = x + m*(long)HQ;
  const float* wrow[NT];
  #pragma unroll
  for (int nt = 0; nt < NT; ++nt)
    wrow[nt] = W + (long)(gn*NT*16 + nt*16 + r)*HQ;
  v8f acc[NT] = {};
  #pragma unroll
  for (int k0 = 0; k0 < HQ; k0 += 32) {
    v16bf a;
    #pragma unroll
    for (int i = 0; i < 16; ++i) {
      int ka = k0 + kmapA(i, hi);
      a[i] = (__bf16)((xrow[ka]-mym)*myr*g[ka] + bb[ka]);
    }
    #pragma unroll
    for (int nt = 0; nt < NT; ++nt) {
      v16bf b;
      #pragma unroll
      for (int i = 0; i < 16; ++i) b[i] = (__bf16)wrow[nt][k0 + kmapB(i, hi)];
      acc[nt] = __builtin_amdgcn_wmma_f32_16x16x32_bf16(false, a, false, b, (short)0, acc[nt], false, false);
    }
  }
  #pragma unroll
  for (int nt = 0; nt < NT; ++nt) {
    float* orow = xz + (tm*16 + hi*8)*(long)XZLD + gn*NT*16 + nt*16 + r;
    #pragma unroll
    for (int e = 0; e < 8; ++e) orow[(long)e*XZLD] = acc[nt][e];
  }
}

// ===== x_proj GEMM, causal depthwise conv + SiLU fused on A:  xdbl = silu(conv(u)) @ xproj_w^T =====
// One wave owns 1 M-tile x all 3 (padded) N-tiles.
__global__ __launch_bounds__(256)
void k_xproj(const float* __restrict__ xz, const float* __restrict__ cw,
             const float* __restrict__ cb, const float* __restrict__ W,
             float* __restrict__ xdbl) {
  const int NT = 3;
  long wid = (long)blockIdx.x*8 + (threadIdx.x >> 5);
  const long total = (long)(MROWS/16);
  if (wid >= total) return;
  const int lane = threadIdx.x & 31, r = lane & 15, hi = lane >> 4;
  const long tm = wid;
  const long m = tm*16 + r;
  const int  t = (int)(m % TQ);
  const float* urow = xz + m*(long)XZLD;                   // u occupies cols [0,192)
  v8f acc[NT] = {};
  #pragma unroll
  for (int k0 = 0; k0 < DIq; k0 += 32) {
    v16bf a;
    #pragma unroll
    for (int i = 0; i < 16; ++i) {
      int d = k0 + kmapA(i, hi);
      float s = cb[d];
      #pragma unroll
      for (int j = 0; j < DCq; ++j)
        if (t - 3 + j >= 0) s += cw[d*DCq+j] * urow[(long)(j-3)*XZLD + d];
      a[i] = (__bf16)siluf(s);
    }
    #pragma unroll
    for (int nt = 0; nt < NT; ++nt) {
      const int n = nt*16 + r;
      v16bf b;
      #pragma unroll
      for (int i = 0; i < 16; ++i)
        b[i] = (n < DRq + 2*NST) ? (__bf16)W[(long)n*DIq + k0 + kmapB(i, hi)] : (__bf16)0.f;
      acc[nt] = __builtin_amdgcn_wmma_f32_16x16x32_bf16(false, a, false, b, (short)0, acc[nt], false, false);
    }
  }
  #pragma unroll
  for (int nt = 0; nt < NT; ++nt) {
    float* orow = xdbl + (tm*16 + hi*8)*(long)XDLD + nt*16 + r;
    #pragma unroll
    for (int e = 0; e < 8; ++e) orow[(long)e*XDLD] = acc[nt][e];
  }
}

// ===== selective scan: dt-proj + softplus + conv window + SSM recurrence + gate, y overwrites u =====
__global__ __launch_bounds__(192)
void k_scan(float* __restrict__ xz, const float* __restrict__ xdbl,
            const float* __restrict__ cw, const float* __restrict__ cb,
            const float* __restrict__ dtw, const float* __restrict__ dtb,
            const float* __restrict__ Alog, const float* __restrict__ Dp) {
  __shared__ float row[2][40];
  const int bf = blockIdx.x;
  const int d  = threadIdx.x;                  // 0..191
  float Ad[NST];
  #pragma unroll
  for (int n = 0; n < NST; ++n) Ad[n] = -__expf(Alog[d*NST+n]);
  float cwj[DCq];
  #pragma unroll
  for (int j = 0; j < DCq; ++j) cwj[j] = cw[d*DCq+j];
  const float cbv = cb[d];
  float dw[DRq];
  #pragma unroll
  for (int q = 0; q < DRq; ++q) dw[q] = dtw[d*DRq+q];
  const float db = dtb[d], Dd = Dp[d];
  float h[NST];
  #pragma unroll
  for (int n = 0; n < NST; ++n) h[n] = 0.f;
  float u1 = 0.f, u2 = 0.f, u3 = 0.f;          // rolling causal-conv window
  const long base = (long)bf * TQ;
  for (int t = 0; t < TQ; ++t) {
    const int p = t & 1;
    if (d < DRq + 2*NST) row[p][d] = xdbl[(base+t)*XDLD + d];
    __syncthreads();
    float uraw = xz[(base+t)*XZLD + d];
    float z    = xz[(base+t)*XZLD + DIq + d];
    if (t + 1 < TQ) {                           // overlap next step's fetch with recurrence
      __builtin_prefetch(&xz[(base+t+1)*XZLD + d], 0, 3);
      __builtin_prefetch(&xz[(base+t+1)*XZLD + DIq + d], 0, 3);
    }
    float s = cbv + cwj[0]*u3 + cwj[1]*u2 + cwj[2]*u1 + cwj[3]*uraw;
    float uc = siluf(s);
    float dtr = db;
    #pragma unroll
    for (int q = 0; q < DRq; ++q) dtr += dw[q]*row[p][q];
    float dt = (dtr > 20.f) ? dtr : log1pf(__expf(dtr));
    float du = dt * uc;
    float y = 0.f;
    #pragma unroll
    for (int n = 0; n < NST; ++n) {
      float dA = __expf(dt * Ad[n]);
      h[n] = h[n]*dA + du*row[p][DRq+n];
      y += h[n]*row[p][DRq+NST+n];
    }
    y += uc * Dd;
    y *= siluf(z);
    xz[(base+t)*XZLD + d] = y;                 // u[t] fully consumed -> overwrite with y
    u3 = u2; u2 = u1; u1 = uraw;
  }
}

// ===== out_proj GEMM with fused residual: out += y @ out_w^T  (K=192, N=96) =====
// One wave owns 1 M-tile x 3 N-tiles (2 groups cover N=96).
__global__ __launch_bounds__(256)
void k_outproj(const float* __restrict__ xz, const float* __restrict__ W,
               float* __restrict__ out) {
  const int NT = 3, NGRP = HQ/16/NT;                      // 2 groups
  long wid = (long)blockIdx.x*8 + (threadIdx.x >> 5);
  const long total = (long)(MROWS/16)*NGRP;
  if (wid >= total) return;
  const int lane = threadIdx.x & 31, r = lane & 15, hi = lane >> 4;
  const int  gn = (int)(wid % NGRP);
  const long tm = wid / NGRP;
  const long m = tm*16 + r;
  const float* yrow = xz + m*(long)XZLD;
  const float* wrow[NT];
  #pragma unroll
  for (int nt = 0; nt < NT; ++nt)
    wrow[nt] = W + (long)(gn*NT*16 + nt*16 + r)*DIq;
  v8f acc[NT] = {};
  #pragma unroll
  for (int k0 = 0; k0 < DIq; k0 += 32) {
    v16bf a;
    #pragma unroll
    for (int i = 0; i < 16; ++i) a[i] = (__bf16)yrow[k0 + kmapA(i, hi)];
    #pragma unroll
    for (int nt = 0; nt < NT; ++nt) {
      v16bf b;
      #pragma unroll
      for (int i = 0; i < 16; ++i) b[i] = (__bf16)wrow[nt][k0 + kmapB(i, hi)];
      acc[nt] = __builtin_amdgcn_wmma_f32_16x16x32_bf16(false, a, false, b, (short)0, acc[nt], false, false);
    }
  }
  #pragma unroll
  for (int nt = 0; nt < NT; ++nt) {
    float* orow = out + (tm*16 + hi*8)*(long)HQ + gn*NT*16 + nt*16 + r;
    #pragma unroll
    for (int e = 0; e < 8; ++e) orow[(long)e*HQ] += acc[nt][e];
  }
}

// ================= host launcher =================
extern "C" void kernel_launch(void* const* d_in, const int* in_sizes, int n_in,
                              void* d_out, int out_size, void* d_ws, size_t ws_size,
                              hipStream_t stream) {
  (void)in_sizes; (void)n_in; (void)out_size; (void)ws_size;
  const float* x = (const float*)d_in[0];
  // dict order: x[0], fc1[1..5], fc2[6..10], full_ln[11,12], sq[13,14], full[15,16], unsq[17,18], m1[19..29], m2[30..40]
  float* out = (float*)d_out;
  float* ws  = (float*)d_ws;
  float* xz   = ws;                               // 258000*384
  float* xdbl = xz + (size_t)MROWS*XZLD;          // 258000*48
  float* mu   = xdbl + (size_t)MROWS*XDLD;        // 258000
  float* rs   = mu + MROWS;                       // 258000

  size_t smem1 = (size_t)(2*FQ*HQ + 2*DSq*FQ) * sizeof(float);   // ~107 KB (within 320 KB/WGP)
  k_stage1<<<BQ*TQ, 256, smem1, stream>>>(
      x, out,
      (const float*)d_in[1], (const float*)d_in[2], (const float*)d_in[3], (const float*)d_in[4], (const float*)d_in[5],
      (const float*)d_in[6], (const float*)d_in[7], (const float*)d_in[8], (const float*)d_in[9], (const float*)d_in[10],
      (const float*)d_in[11], (const float*)d_in[12],
      (const float*)d_in[13], (const float*)d_in[14],
      (const float*)d_in[15], (const float*)d_in[16],
      (const float*)d_in[17], (const float*)d_in[18]);

  const float* mp[2][11];
  for (int i = 0; i < 11; ++i) {
    mp[0][i] = (const float*)d_in[19+i];
    mp[1][i] = (const float*)d_in[30+i];
  }
  // per-param order: 0 ln_g, 1 ln_b, 2 in_w, 3 conv_w, 4 conv_b, 5 xproj_w,
  //                  6 dt_w, 7 dt_b, 8 Alog, 9 D, 10 out_w
  const long inprojWaves  = (long)(MROWS/16)*(XZLD/16/4);   // 96750
  const long xprojWaves   = (long)(MROWS/16);               // 16125
  const long outprojWaves = (long)(MROWS/16)*(HQ/16/3);     // 32250
  for (int sblk = 0; sblk < 2; ++sblk) {
    k_stats<<<MROWS/8, 256, 0, stream>>>(out, mu, rs);
    k_inproj<<<(int)((inprojWaves + 7)/8), 256, 0, stream>>>(
        out, mu, rs, mp[sblk][0], mp[sblk][1], mp[sblk][2], xz);
    k_xproj<<<(int)((xprojWaves + 7)/8), 256, 0, stream>>>(
        xz, mp[sblk][3], mp[sblk][4], mp[sblk][5], xdbl);
    k_scan<<<BQ*FQ, 192, 0, stream>>>(
        xz, xdbl, mp[sblk][3], mp[sblk][4], mp[sblk][6], mp[sblk][7], mp[sblk][8], mp[sblk][9]);
    k_outproj<<<(int)((outprojWaves + 7)/8), 256, 0, stream>>>(xz, mp[sblk][10], out);
  }
}
